// GATNetWithAttention_10660108829398
// MI455X (gfx1250) — matrix operand order repaired and checked by reading, not compiled, not run on hardware
//
#include <hip/hip_runtime.h>
#include <hip/hip_bf16.h>
#include <math.h>

// ---------------------------------------------------------------------------
// GATNetWithAttention for MI455X (gfx1250, wave32).
// Dense GEMMs -> v_wmma_f32_16x16x32_bf16 (bf16 in, f32 accumulate).
// Graph softmax/aggregation -> L2-resident atomics (all per-node tensors
// fit in the 192MB L2, so the random src/dst traffic never leaves the die).
// ---------------------------------------------------------------------------

#define NNODES 50000
#define NEDGES 800000
#define IN_F   128
#define H1_    8
#define D1_    32
#define F1_    256      // H1_*D1_
#define F2_    128
#define FC1_   512
#define NEG_SLOPE 0.2f

typedef __attribute__((ext_vector_type(16))) __bf16 v16bf;
typedef __attribute__((ext_vector_type(8)))  __bf16 v8bf;
typedef __attribute__((ext_vector_type(8)))  float  v8f;

// ----- order-preserving float<->uint encoding (for atomic segment-max) -----
static __device__ __forceinline__ unsigned enc_ordered(float f) {
  unsigned u = __float_as_uint(f);
  return (u & 0x80000000u) ? ~u : (u | 0x80000000u);
}
static __device__ __forceinline__ float dec_ordered(unsigned e) {
  unsigned u = (e & 0x80000000u) ? (e & 0x7FFFFFFFu) : ~e;
  return __uint_as_float(u);
}
static __device__ __forceinline__ float leaky(float x) {
  return x > 0.0f ? x : NEG_SLOPE * x;
}

// ---------------------------------------------------------------------------
// Utility kernels
// ---------------------------------------------------------------------------
__global__ void fill_u32(unsigned* __restrict__ p, unsigned v, int n) {
  int i = blockIdx.x * blockDim.x + threadIdx.x;
  if (i < n) p[i] = v;
}

__global__ void f32_to_bf16(const float* __restrict__ s, __bf16* __restrict__ d, int n) {
  int i = blockIdx.x * blockDim.x + threadIdx.x;
  if (i < n) d[i] = (__bf16)s[i];
}

// B[K,Ncol] f32 (row-major)  ->  Bt[Ncol,K] bf16  (i.e. column-major of B)
__global__ void transpose_to_bf16(const float* __restrict__ s, __bf16* __restrict__ d,
                                  int K, int Ncol) {
  int i = blockIdx.x * blockDim.x + threadIdx.x;
  if (i >= K * Ncol) return;
  int k = i / Ncol, n = i % Ncol;
  d[(size_t)n * K + k] = (__bf16)s[i];
}

__global__ void init_out(float* __restrict__ out, const float* __restrict__ fcb2, int n) {
  int i = blockIdx.x * blockDim.x + threadIdx.x;
  if (i < n) out[i] = fcb2[0];
}

// ---------------------------------------------------------------------------
// WMMA GEMM: C[M,Ncol] = A[M,K] * B[K,Ncol], A row-major bf16, Bt = B^T bf16.
// One wave = one 16x16 tile of C. ISA lane layouts (CDNA5 ISA):
//   A (16x32 bf16): lane l holds row M=l&15; elems e<8 -> K = 8*(l>>4)+e,
//                   e>=8 -> K = 16 + 8*(l>>4) + (e-8)   (two contiguous runs)
//   B (32x16 bf16): lane l holds col N=l&15; code uses elem e -> K = 16*(l>>4)+e for both B and A
//   C (16x16 f32):  vgpr r, lane l -> row r+8*(l>>4), col l&15
// ---------------------------------------------------------------------------
__global__ void wmma_gemm_bf16(const __bf16* __restrict__ A,
                               const __bf16* __restrict__ Bt,
                               float* __restrict__ C,
                               int M, int K, int Ncol) {
  int wave = (int)((blockIdx.x * blockDim.x + threadIdx.x) >> 5);
  int lane = threadIdx.x & 31;
  int tilesN = Ncol >> 4;
  int tileM = wave / tilesN;
  int tileN = wave % tilesN;
  if (tileM >= (M >> 4)) return;

  int half = lane >> 4;      // 0: lanes 0-15, 1: lanes 16-31
  int lmod = lane & 15;

  const __bf16* arow = A  + (size_t)(tileM * 16 + lmod) * K + half * 16;
  const __bf16* bcol = Bt + (size_t)(tileN * 16 + lmod) * K + half * 16;

  v8f c = {};
  for (int kk = 0; kk < K; kk += 32) {
    v8bf alo = *(const v8bf*)(arow + kk);        // K = kk + 16*half + 0..7
    v8bf ahi = *(const v8bf*)(arow + kk + 8);    // K = kk + 16*half + 8..15
    v16bf a, b;
#pragma unroll
    for (int i = 0; i < 8; ++i) { a[i] = alo[i]; a[i + 8] = ahi[i]; }
    b = *(const v16bf*)(bcol + kk);              // K = kk + 16*half + 0..15
    c = __builtin_amdgcn_wmma_f32_16x16x32_bf16(false, a, false, b,
                                                (short)0, c, false, false);
  }
  float* crow = C + (size_t)(tileM * 16 + half * 8) * Ncol + tileN * 16 + lmod;
#pragma unroll
  for (int r = 0; r < 8; ++r) crow[(size_t)r * Ncol] = c[r];
}

// ---------------------------------------------------------------------------
// Fused FC head: h = relu(X @ fcW1 + fcb1); out[node] += sum_j h[j]*fcW2[j]
// WMMA tiles of FC1, epilogue reduces the 16 tile-columns across lanes and
// atomically accumulates into out (pre-initialized with fcb2).
// ---------------------------------------------------------------------------
__global__ void wmma_fc_fused(const __bf16* __restrict__ X,     // [M,K] bf16
                              const __bf16* __restrict__ W1t,   // [Ncol,K] bf16
                              const float* __restrict__ fcb1,   // [Ncol]
                              const float* __restrict__ fcW2,   // [Ncol]
                              float* __restrict__ out,          // [M]
                              int M, int K, int Ncol) {
  int wave = (int)((blockIdx.x * blockDim.x + threadIdx.x) >> 5);
  int lane = threadIdx.x & 31;
  int tilesN = Ncol >> 4;
  int tileM = wave / tilesN;
  int tileN = wave % tilesN;
  if (tileM >= (M >> 4)) return;

  int half = lane >> 4;
  int lmod = lane & 15;

  const __bf16* arow = X   + (size_t)(tileM * 16 + lmod) * K + half * 16;
  const __bf16* bcol = W1t + (size_t)(tileN * 16 + lmod) * K + half * 16;

  v8f c = {};
  for (int kk = 0; kk < K; kk += 32) {
    v8bf alo = *(const v8bf*)(arow + kk);
    v8bf ahi = *(const v8bf*)(arow + kk + 8);
    v16bf a, b;
#pragma unroll
    for (int i = 0; i < 8; ++i) { a[i] = alo[i]; a[i + 8] = ahi[i]; }
    b = *(const v16bf*)(bcol + kk);
    c = __builtin_amdgcn_wmma_f32_16x16x32_bf16(false, a, false, b,
                                                (short)0, c, false, false);
  }

  int col = tileN * 16 + lmod;        // this lane's FC1 column
  float b1v = fcb1[col];
  float w2v = fcW2[col];
#pragma unroll
  for (int r = 0; r < 8; ++r) {
    float h = c[r] + b1v;
    h = h > 0.0f ? h : 0.0f;
    float v = h * w2v;
    // reduce across the 16 lanes that hold this row's tile columns
    v += __shfl_xor(v, 1);
    v += __shfl_xor(v, 2);
    v += __shfl_xor(v, 4);
    v += __shfl_xor(v, 8);
    if (lmod == 0)
      atomicAdd(&out[tileM * 16 + r + 8 * half], v);
  }
}

// ---------------------------------------------------------------------------
// Attention coefficients: as[n,h] = sum_d H[n,h*D+d]*att_src[h,d] (same for ad)
// One block per node; blockDim = heads*dim (256 or 128).
// ---------------------------------------------------------------------------
__global__ void attn_coef(const float* __restrict__ H,
                          const float* __restrict__ att_s,
                          const float* __restrict__ att_d,
                          float* __restrict__ as_, float* __restrict__ ad_,
                          int heads, int dim) {
  int n = blockIdx.x;
  int t = threadIdx.x;
  int F = heads * dim;
  float hv = H[(size_t)n * F + t];
  __shared__ float ss[256];
  __shared__ float sd[256];
  ss[t] = hv * att_s[t];
  sd[t] = hv * att_d[t];
  __syncthreads();
  for (int s = dim >> 1; s > 0; s >>= 1) {
    if ((t % dim) < s) { ss[t] += ss[t + s]; sd[t] += sd[t + s]; }
    __syncthreads();
  }
  if ((t % dim) == 0) {
    int h = t / dim;
    as_[(size_t)n * heads + h] = ss[t];
    ad_[(size_t)n * heads + h] = sd[t];
  }
}

// ---------------------------------------------------------------------------
// Segment softmax over dst, four passes (all L2-resident).
// ---------------------------------------------------------------------------
__global__ void edge_max(const int* __restrict__ src, const int* __restrict__ dst,
                         const float* __restrict__ as_, const float* __restrict__ ad_,
                         unsigned* __restrict__ m, int E, int Hh) {
  int idx = blockIdx.x * blockDim.x + threadIdx.x;
  if (idx >= E * Hh) return;
  int e = idx / Hh, h = idx % Hh;
  int s = src[e], d = dst[e];
  float x = leaky(as_[(size_t)s * Hh + h] + ad_[(size_t)d * Hh + h]);
  atomicMax(&m[(size_t)d * Hh + h], enc_ordered(x));
}

__global__ void decode_max(unsigned* __restrict__ m, int n) {  // in-place -> float
  int i = blockIdx.x * blockDim.x + threadIdx.x;
  if (i >= n) return;
  float f = dec_ordered(m[i]);
  if (!isfinite(f)) f = 0.0f;   // empty segments, like reference
  ((float*)m)[i] = f;
}

__global__ void edge_exp(const int* __restrict__ src, const int* __restrict__ dst,
                         const float* __restrict__ as_, const float* __restrict__ ad_,
                         const float* __restrict__ m, float* __restrict__ albuf,
                         float* __restrict__ den, int E, int Hh) {
  int idx = blockIdx.x * blockDim.x + threadIdx.x;
  if (idx >= E * Hh) return;
  int e = idx / Hh, h = idx % Hh;
  int s = src[e], d = dst[e];
  float x  = leaky(as_[(size_t)s * Hh + h] + ad_[(size_t)d * Hh + h]);
  float ex = __expf(x - m[(size_t)d * Hh + h]);
  albuf[idx] = ex;
  atomicAdd(&den[(size_t)d * Hh + h], ex);
}

__global__ void edge_normalize(const int* __restrict__ dst,
                               float* __restrict__ albuf,
                               const float* __restrict__ den, int E, int Hh) {
  int idx = blockIdx.x * blockDim.x + threadIdx.x;
  if (idx >= E * Hh) return;
  int e = idx / Hh, h = idx % Hh;
  albuf[idx] = albuf[idx] / (den[(size_t)dst[e] * Hh + h] + 1e-16f);
}

// One block per edge (blockDim = heads*dim): agg[dst,f] += hfeat[src,f]*alpha
__global__ void edge_scatter(const int* __restrict__ src, const int* __restrict__ dst,
                             const float* __restrict__ hfeat,
                             const float* __restrict__ albuf,
                             float* __restrict__ agg, int Hh, int dim) {
  int e = blockIdx.x;
  int t = threadIdx.x;
  int F = Hh * dim;
  int s = src[e], d = dst[e];
  float alpha = albuf[(size_t)e * Hh + (t / dim)];
  atomicAdd(&agg[(size_t)d * F + t], hfeat[(size_t)s * F + t] * alpha);
}

// in-place: x = act(x + bias[f]);  mode 0 = ELU, 1 = ReLU
__global__ void bias_act(float* __restrict__ x, const float* __restrict__ b,
                         int F, int total, int mode) {
  int i = blockIdx.x * blockDim.x + threadIdx.x;
  if (i >= total) return;
  float v = x[i] + b[i % F];
  if (mode) v = v > 0.0f ? v : 0.0f;
  else      v = v > 0.0f ? v : expm1f(v);
  x[i] = v;
}

// ---------------------------------------------------------------------------
// Host orchestration
// ---------------------------------------------------------------------------
static inline int ceil_div(int a, int b) { return (a + b - 1) / b; }

extern "C" void kernel_launch(void* const* d_in, const int* in_sizes, int n_in,
                              void* d_out, int out_size, void* d_ws, size_t ws_size,
                              hipStream_t stream) {
  const float* x        = (const float*)d_in[0];
  const int*   eidx     = (const int*)  d_in[1];
  const float* W1       = (const float*)d_in[2];
  const float* att_src1 = (const float*)d_in[3];
  const float* att_dst1 = (const float*)d_in[4];
  const float* b1       = (const float*)d_in[5];
  const float* W2       = (const float*)d_in[6];
  const float* att_src2 = (const float*)d_in[7];
  const float* att_dst2 = (const float*)d_in[8];
  const float* b2       = (const float*)d_in[9];
  const float* fcW1     = (const float*)d_in[10];
  const float* fcb1     = (const float*)d_in[11];
  const float* fcW2     = (const float*)d_in[12];
  const float* fcb2     = (const float*)d_in[13];

  const int* src = eidx;
  const int* dst = eidx + NEDGES;

  float* out_node = (float*)d_out;                         // [N]
  float* alpha1   = (float*)d_out + NNODES;                // [E,8]
  float* alpha2   = alpha1 + (size_t)NEDGES * H1_;         // [E,1]

  // workspace carve-up (256B aligned)
  char* w = (char*)d_ws;
  auto alloc = [&](size_t bytes) -> void* {
    void* p = (void*)w;
    w += (bytes + 255) & ~(size_t)255;
    return p;
  };
  __bf16* xbf   = (__bf16*)alloc((size_t)NNODES * IN_F * 2);
  __bf16* w1t   = (__bf16*)alloc((size_t)F1_ * IN_F * 2);
  float*  h1    = (float*) alloc((size_t)NNODES * F1_ * 4);
  float*  as1   = (float*) alloc((size_t)NNODES * H1_ * 4);
  float*  ad1   = (float*) alloc((size_t)NNODES * H1_ * 4);
  unsigned* m1  = (unsigned*)alloc((size_t)NNODES * H1_ * 4);
  float*  den1  = (float*) alloc((size_t)NNODES * H1_ * 4);
  float*  agg1  = (float*) alloc((size_t)NNODES * F1_ * 4);   // becomes x1
  __bf16* x1bf  = (__bf16*)alloc((size_t)NNODES * F1_ * 2);
  __bf16* w2t   = (__bf16*)alloc((size_t)F2_ * F1_ * 2);
  float*  h2    = (float*) alloc((size_t)NNODES * F2_ * 4);
  float*  as2   = (float*) alloc((size_t)NNODES * 4);
  float*  ad2   = (float*) alloc((size_t)NNODES * 4);
  unsigned* m2  = (unsigned*)alloc((size_t)NNODES * 4);
  float*  den2  = (float*) alloc((size_t)NNODES * 4);
  float*  agg2  = (float*) alloc((size_t)NNODES * F2_ * 4);   // becomes x2
  __bf16* x2bf  = (__bf16*)alloc((size_t)NNODES * F2_ * 2);
  __bf16* fw1t  = (__bf16*)alloc((size_t)FC1_ * F2_ * 2);

  const unsigned ENC_NEG_INF = 0x007FFFFFu;  // enc_ordered(-inf)
  const int T = 256;

  // ---------------- layer 1 ----------------
  f32_to_bf16<<<ceil_div(NNODES * IN_F, T), T, 0, stream>>>(x, xbf, NNODES * IN_F);
  transpose_to_bf16<<<ceil_div(IN_F * F1_, T), T, 0, stream>>>(W1, w1t, IN_F, F1_);
  {
    int tiles = (NNODES / 16) * (F1_ / 16);
    wmma_gemm_bf16<<<ceil_div(tiles, 8), 256, 0, stream>>>(xbf, w1t, h1, NNODES, IN_F, F1_);
  }
  attn_coef<<<NNODES, F1_, 0, stream>>>(h1, att_src1, att_dst1, as1, ad1, H1_, D1_);

  fill_u32<<<ceil_div(NNODES * H1_, T), T, 0, stream>>>(m1, ENC_NEG_INF, NNODES * H1_);
  fill_u32<<<ceil_div(NNODES * H1_, T), T, 0, stream>>>((unsigned*)den1, 0u, NNODES * H1_);
  fill_u32<<<ceil_div(NNODES * F1_, T), T, 0, stream>>>((unsigned*)agg1, 0u, NNODES * F1_);

  edge_max<<<ceil_div(NEDGES * H1_, T), T, 0, stream>>>(src, dst, as1, ad1, m1, NEDGES, H1_);
  decode_max<<<ceil_div(NNODES * H1_, T), T, 0, stream>>>(m1, NNODES * H1_);
  edge_exp<<<ceil_div(NEDGES * H1_, T), T, 0, stream>>>(src, dst, as1, ad1, (float*)m1,
                                                        alpha1, den1, NEDGES, H1_);
  edge_normalize<<<ceil_div(NEDGES * H1_, T), T, 0, stream>>>(dst, alpha1, den1, NEDGES, H1_);
  edge_scatter<<<NEDGES, F1_, 0, stream>>>(src, dst, h1, alpha1, agg1, H1_, D1_);
  bias_act<<<ceil_div(NNODES * F1_, T), T, 0, stream>>>(agg1, b1, F1_, NNODES * F1_, 0); // ELU

  // ---------------- layer 2 ----------------
  f32_to_bf16<<<ceil_div(NNODES * F1_, T), T, 0, stream>>>(agg1, x1bf, NNODES * F1_);
  transpose_to_bf16<<<ceil_div(F1_ * F2_, T), T, 0, stream>>>(W2, w2t, F1_, F2_);
  {
    int tiles = (NNODES / 16) * (F2_ / 16);
    wmma_gemm_bf16<<<ceil_div(tiles, 8), 256, 0, stream>>>(x1bf, w2t, h2, NNODES, F1_, F2_);
  }
  attn_coef<<<NNODES, F2_, 0, stream>>>(h2, att_src2, att_dst2, as2, ad2, 1, F2_);

  fill_u32<<<ceil_div(NNODES, T), T, 0, stream>>>(m2, ENC_NEG_INF, NNODES);
  fill_u32<<<ceil_div(NNODES, T), T, 0, stream>>>((unsigned*)den2, 0u, NNODES);
  fill_u32<<<ceil_div(NNODES * F2_, T), T, 0, stream>>>((unsigned*)agg2, 0u, NNODES * F2_);

  edge_max<<<ceil_div(NEDGES, T), T, 0, stream>>>(src, dst, as2, ad2, m2, NEDGES, 1);
  decode_max<<<ceil_div(NNODES, T), T, 0, stream>>>(m2, NNODES);
  edge_exp<<<ceil_div(NEDGES, T), T, 0, stream>>>(src, dst, as2, ad2, (float*)m2,
                                                  alpha2, den2, NEDGES, 1);
  edge_normalize<<<ceil_div(NEDGES, T), T, 0, stream>>>(dst, alpha2, den2, NEDGES, 1);
  edge_scatter<<<NEDGES, F2_, 0, stream>>>(src, dst, h2, alpha2, agg2, 1, F2_);
  bias_act<<<ceil_div(NNODES * F2_, T), T, 0, stream>>>(agg2, b2, F2_, NNODES * F2_, 1); // ReLU

  // ---------------- fused FC head ----------------
  f32_to_bf16<<<ceil_div(NNODES * F2_, T), T, 0, stream>>>(agg2, x2bf, NNODES * F2_);
  transpose_to_bf16<<<ceil_div(F2_ * FC1_, T), T, 0, stream>>>(fcW1, fw1t, F2_, FC1_);
  init_out<<<ceil_div(NNODES, T), T, 0, stream>>>(out_node, fcb2, NNODES);
  {
    int tiles = (NNODES / 16) * (FC1_ / 16);
    wmma_fc_fused<<<ceil_div(tiles, 8), 256, 0, stream>>>(x2bf, fw1t, fcb1, fcW2,
                                                          out_node, NNODES, F2_, FC1_);
  }
}
